// KATBlocks_17609365913760
// MI455X (gfx1250) — matrix operand order repaired and checked by reading, not compiled, not run on hardware
//
#include <hip/hip_runtime.h>
#include <hip/hip_bf16.h>

// ---------------- problem constants ----------------
#define BB 2
#define NN 4096
#define MM 256
#define DIMD 512
#define DEPTH 2
#define HEADS 8
#define DHEAD 64
#define INNER 512        // HEADS*DHEAD
#define THREE 1536       // 3*INNER
#define MLPD 2048
#define NEGV -1.0e9f

typedef __attribute__((ext_vector_type(16))) __bf16 v16bf;
typedef __attribute__((ext_vector_type(8)))  float  v8f;
typedef __attribute__((ext_vector_type(4)))  unsigned int v4u;
typedef __attribute__((ext_vector_type(8)))  int v8i;
typedef __attribute__((ext_vector_type(4)))  int v4i;

// ---------------- helpers ----------------
__device__ __forceinline__ __bf16 f2bf(float f) {
    unsigned u = __float_as_uint(f);
    unsigned r = u + 0x7FFFu + ((u >> 16) & 1u);   // round-to-nearest-even
    unsigned short h = (unsigned short)(r >> 16);
    return __builtin_bit_cast(__bf16, h);
}
__device__ __forceinline__ unsigned pack2bf(float a, float b) {
    unsigned ua = __float_as_uint(a); ua = ua + 0x7FFFu + ((ua >> 16) & 1u);
    unsigned ub = __float_as_uint(b); ub = ub + 0x7FFFu + ((ub >> 16) & 1u);
    return (ua >> 16) | (ub & 0xFFFF0000u);
}
__device__ __forceinline__ float geluf(float x) {
    return 0.5f * x * (1.0f + erff(x * 0.70710678118654752f));
}
__device__ __forceinline__ float wred_sum(float v) {
    #pragma unroll
    for (int o = 16; o > 0; o >>= 1) v += __shfl_xor(v, o, 32);
    return v;
}
__device__ __forceinline__ float wred_max(float v) {
    #pragma unroll
    for (int o = 16; o > 0; o >>= 1) v = fmaxf(v, __shfl_xor(v, o, 32));
    return v;
}

// ---- Tensor Data Mover: load a 2D f32 tile [tile rows=128 x 32 cols] into LDS,
// ---- zero-filling rows past rows_valid; LDS rows padded 32+4 dwords.
__device__ __forceinline__ void tdm_load_a_tile(
    const float* gsrc, unsigned lds_byte_off, int rows_valid, int lda)
{
    unsigned long long ga = (unsigned long long)(size_t)gsrc;
    v4u g0;
    g0[0] = 1u;                                        // count=1 (valid user D#)
    g0[1] = lds_byte_off;                              // lds_addr (bytes)
    g0[2] = (unsigned)(ga & 0xFFFFFFFFull);            // global_addr[31:0]
    g0[3] = (unsigned)((ga >> 32) & 0x1FFFFFFull) | (2u << 30);  // addr[56:32], type=2
    unsigned w[8] = {};
    // data_size=4B (code 2), pad_enable, pad_interval=32 dwords (code 4),
    // pad_amount=4 dwords (code 3)
    w[0] = (2u << 16) | (1u << 20) | (4u << 22) | (3u << 25);
    w[1] = (32u & 0xFFFFu) << 16;                      // tensor_dim0[15:0] = 32
    w[2] = ((unsigned)rows_valid & 0xFFFFu) << 16;     // dim0 hi=0 | tensor_dim1 lo
    w[3] = (((unsigned)rows_valid >> 16) & 0xFFFFu) | (32u << 16); // dim1 hi | tile_dim0=32
    w[4] = 128u;                                       // tile_dim1=128, tile_dim2=0
    w[5] = (unsigned)lda;                              // tensor_dim0_stride[31:0]
    v8i g1;
    #pragma unroll
    for (int i = 0; i < 8; ++i) g1[i] = (int)w[i];
    v4i z4 = {};
#if defined(__clang_major__) && (__clang_major__ >= 23)
    v8i z8 = {};
    __builtin_amdgcn_tensor_load_to_lds(g0, g1, z4, z4, z8, 0);
#else
    __builtin_amdgcn_tensor_load_to_lds(g0, g1, z4, z4, 0);
#endif
}

// ---------------- generic WMMA GEMM ----------------
// C[M,N] = act( alpha * A[M,K] x B[K,N] + bias ) + Cres    (f32 in, bf16 WMMA, f32 acc)
// transB: B stored [N,K] row-major.  Requires K%32==0, N%64==0 (true for all calls).
// Batched over blockIdx.z = b*Hdim + h with 64-bit strides.
#define BMT 128
#define BNT 64
#define BKT 32

__global__ __launch_bounds__(256) void gemm_wmma_bf16(
    const float* __restrict__ A, const float* __restrict__ Bm,
    const float* __restrict__ bias, const float* __restrict__ Cres,
    float* __restrict__ C,
    int M, int N, int K, int lda, int ldb, int ldc,
    long long sAb, long long sAh, long long sBb, long long sBh,
    long long sCb, long long sCh,
    int Hdim, int transB, float alpha, int act)
{
    __shared__ float  Asf[BMT][36];        // TDM-staged f32 A tile (32 + 4 pad per row)
    __shared__ __bf16 Bs[BNT][BKT + 8];    // n-major bf16 B tile

    const int z  = blockIdx.z;
    const int bb = z / Hdim, hh = z % Hdim;
    A  += (size_t)bb * sAb + (size_t)hh * sAh;
    Bm += (size_t)bb * sBb + (size_t)hh * sBh;
    C  += (size_t)bb * sCb + (size_t)hh * sCh;
    if (Cres) Cres += (size_t)bb * sCb + (size_t)hh * sCh;

    const int bm   = blockIdx.y * BMT;
    const int bn   = blockIdx.x * BNT;
    const int t    = threadIdx.x;
    const int wave = t >> 5;
    const int lane = t & 31;
    const int rows_valid = M - bm;                 // TDM zero-fills rows beyond this
    const unsigned lds_a = (unsigned)(size_t)(void*)&Asf[0][0];

    v8f acc[4] = {};

    for (int kk = 0; kk < K; kk += BKT) {
        __syncthreads();   // previous tile fully consumed before overwrite

        // ---- A tile via Tensor Data Mover (issued by wave 0 only) ----
        if (wave == 0) {
            tdm_load_a_tile(A + (size_t)bm * lda + kk, lds_a, rows_valid, lda);
        }

        // ---- stage B tile (64 n x 32 k), n-major, vectorized (N%64==0) ----
        if (transB) {
            #pragma unroll
            for (int i = 0; i < 2; ++i) {
                int e = t + i * 256;               // 0..511
                int n = e >> 3;                    // 64 rows
                int kc = (e & 7) * 4;              // 0,4,...,28
                const float4 v = *(const float4*)&Bm[(size_t)(bn + n) * ldb + (kk + kc)];
                uint2 p; p.x = pack2bf(v.x, v.y); p.y = pack2bf(v.z, v.w);
                *(uint2*)&Bs[n][kc] = p;
            }
        } else {
            #pragma unroll
            for (int i = 0; i < 2; ++i) {
                int e = t + i * 256;
                int k = e >> 4;                    // 32 k-rows
                int nc = (e & 15) * 4;             // 0,4,...,60
                const float4 v = *(const float4*)&Bm[(size_t)(kk + k) * ldb + (bn + nc)];
                Bs[nc + 0][k] = f2bf(v.x);
                Bs[nc + 1][k] = f2bf(v.y);
                Bs[nc + 2][k] = f2bf(v.z);
                Bs[nc + 3][k] = f2bf(v.w);
            }
        }

        if (kk + BKT < K) {    // cache hint for next B tile (global_prefetch_b8)
            __builtin_prefetch(transB ? &Bm[(size_t)(bn + (t >> 2)) * ldb + kk + BKT]
                                      : &Bm[(size_t)(kk + BKT + (t >> 3)) * ldb + bn],
                               0, 1);
        }

        if (wave == 0) __builtin_amdgcn_s_wait_tensorcnt(0);
        __syncthreads();

        // ---- build A fragment (ISA 16-bit A 16x32 layout), cvt f32->bf16 ----
        const int m    = lane & 15;
        const int koff = (lane < 16) ? 0 : 8;
        v16bf afrag;
        {
            const float* a0 = &Asf[wave * 16 + m][koff];
            const float* a1 = &Asf[wave * 16 + m][16 + koff];
            #pragma unroll
            for (int i = 0; i < 8; ++i) { afrag[i] = f2bf(a0[i]); afrag[i + 8] = f2bf(a1[i]); }
        }
        const int kb = (lane < 16) ? 0 : 8;
        #pragma unroll
        for (int tn = 0; tn < 4; ++tn) {
            const __bf16* q = &Bs[tn * 16 + (lane & 15)][kb];
            v16bf bfrag;
            #pragma unroll
            for (int i = 0; i < 16; ++i) bfrag[i] = q[(i < 8) ? i : (i + 8)];
            acc[tn] = __builtin_amdgcn_wmma_f32_16x16x32_bf16(
                false, afrag, false, bfrag, (short)0, acc[tn], false, false);
        }
    }

    // ---- epilogue (ISA 32-bit C/D 16x16 layout) ----
    #pragma unroll
    for (int tn = 0; tn < 4; ++tn) {
        #pragma unroll
        for (int r = 0; r < 8; ++r) {
            int ml = (lane < 16) ? r : (r + 8);
            int gm = bm + wave * 16 + ml;
            int gn = bn + tn * 16 + (lane & 15);
            if (gm < M && gn < N) {
                float v = acc[tn][r] * alpha;
                if (bias) v += bias[gn];
                if (act == 1) v = geluf(v);
                if (Cres) v += Cres[(size_t)gm * ldc + gn];
                C[(size_t)gm * ldc + gn] = v;
            }
        }
    }
}

// ---------------- LayerNorm: one wave per row of 512 ----------------
__global__ __launch_bounds__(256) void ln_kernel(
    const float* __restrict__ in, const float* __restrict__ g,
    const float* __restrict__ b, float* __restrict__ out, int rows)
{
    int w = blockIdx.x * 8 + (threadIdx.x >> 5);
    int lane = threadIdx.x & 31;
    if (w >= rows) return;
    const float* p = in + (size_t)w * DIMD;
    float v[16]; float s = 0.0f;
    #pragma unroll
    for (int i = 0; i < 16; ++i) { v[i] = p[lane + i * 32]; s += v[i]; }
    s = wred_sum(s);
    float mu = s * (1.0f / DIMD);
    float q = 0.0f;
    #pragma unroll
    for (int i = 0; i < 16; ++i) { float d = v[i] - mu; q += d * d; }
    q = wred_sum(q) * (1.0f / DIMD);
    float inv = rsqrtf(q + 1e-5f);
    float* o = out + (size_t)w * DIMD;
    #pragma unroll
    for (int i = 0; i < 16; ++i) {
        int c = lane + i * 32;
        o[c] = (v[i] - mu) * inv * g[c] + b[c];
    }
}

// ------- softmax over dots rows (len M=256), wave/row; writes attn in place,
// ------- writes attn_sharp (transposed) into amaps output -------
__global__ __launch_bounds__(256) void softmax_attn(
    float* __restrict__ S1, const float* __restrict__ rd,
    const float* __restrict__ mask, const float* __restrict__ kmask,
    float* __restrict__ amaps)
{
    long long w = (long long)blockIdx.x * 8 + (threadIdx.x >> 5);
    int lane = threadIdx.x & 31;
    const long long total = (long long)BB * HEADS * NN;
    if (w >= total) return;
    int b = (int)(w / (HEADS * NN));
    int rem = (int)(w % (HEADS * NN));
    int h = rem / NN, i = rem % NN;
    float* row = S1 + ((size_t)(b * HEADS + h) * NN + i) * MM;
    float mrow = mask[(size_t)b * NN + i];
    float d[8]; float mx = -3.0e38f;
    #pragma unroll
    for (int u = 0; u < 8; ++u) {
        int j = lane + u * 32;
        float x = row[j];
        if (mrow * kmask[(size_t)b * MM + j] < 0.5f) x = NEGV;
        d[u] = x; mx = fmaxf(mx, x);
    }
    mx = wred_max(mx);
    float e1[8], e2[8]; float s1 = 0.0f, s2 = 0.0f;
    #pragma unroll
    for (int u = 0; u < 8; ++u) {
        e1[u] = __expf(d[u] - mx);           s1 += e1[u];
        e2[u] = __expf(24.0f * (d[u] - mx)); s2 += e2[u];
    }
    s1 = wred_sum(s1); s2 = wred_sum(s2);
    float r1 = 1.0f / s1, r2 = 1.0f / s2;
    #pragma unroll
    for (int u = 0; u < 8; ++u) {
        int j = lane + u * 32;
        float rv = rd[((size_t)b * MM + j) * NN + i];     // krd_t[b,i,j]
        row[j] = e1[u] * r1 * rv;
        amaps[(((size_t)b * HEADS + h) * MM + j) * NN + i] = e2[u] * r2 * rv;
    }
}

// ------- softmax over k_dots rows (len N=4096), block/row, * krd -------
__global__ __launch_bounds__(256) void softmax_kattn(
    float* __restrict__ S1, const float* __restrict__ rd,
    const float* __restrict__ mask, const float* __restrict__ kmask)
{
    __shared__ float red[256];
    int rowid = blockIdx.x;                  // over B*H*M
    int b = rowid / (HEADS * MM);
    int rem = rowid % (HEADS * MM);
    int h = rem / MM, i = rem % MM;
    float* row = S1 + ((size_t)(b * HEADS + h) * MM + i) * NN;
    float km = kmask[(size_t)b * MM + i];
    int t = threadIdx.x;
    float d[16]; float mx = -3.0e38f;
    #pragma unroll
    for (int u = 0; u < 16; ++u) {
        int j = t + u * 256;
        float x = row[j];
        if (mask[(size_t)b * NN + j] * km < 0.5f) x = NEGV;
        d[u] = x; mx = fmaxf(mx, x);
    }
    red[t] = mx; __syncthreads();
    for (int o = 128; o > 0; o >>= 1) { if (t < o) red[t] = fmaxf(red[t], red[t + o]); __syncthreads(); }
    mx = red[0]; __syncthreads();
    float e[16]; float s = 0.0f;
    #pragma unroll
    for (int u = 0; u < 16; ++u) { e[u] = __expf(d[u] - mx); s += e[u]; }
    red[t] = s; __syncthreads();
    for (int o = 128; o > 0; o >>= 1) { if (t < o) red[t] += red[t + o]; __syncthreads(); }
    s = red[0];
    float r = 1.0f / s;
    #pragma unroll
    for (int u = 0; u < 16; ++u) {
        int j = t + u * 256;
        row[j] = e[u] * r * rd[((size_t)b * MM + i) * NN + j];   // krd[b,i,j]
    }
}

// ------- cls softmax rows (len M=256), wave/row, no rd multiply -------
__global__ __launch_bounds__(256) void softmax_c(
    float* __restrict__ cd, const float* __restrict__ mask,
    const float* __restrict__ kmask)
{
    int w = blockIdx.x * 8 + (threadIdx.x >> 5);
    int lane = threadIdx.x & 31;
    if (w >= BB * HEADS) return;
    int b = w / HEADS;
    float* row = cd + (size_t)w * MM;
    float m0 = mask[(size_t)b * NN + 0];
    float d[8]; float mx = -3.0e38f;
    #pragma unroll
    for (int u = 0; u < 8; ++u) {
        int j = lane + u * 32;
        float x = row[j];
        if (m0 * kmask[(size_t)b * MM + j] < 0.5f) x = NEGV;
        d[u] = x; mx = fmaxf(mx, x);
    }
    mx = wred_max(mx);
    float e[8]; float s = 0.0f;
    #pragma unroll
    for (int u = 0; u < 8; ++u) { e[u] = __expf(d[u] - mx); s += e[u]; }
    s = wred_sum(s);
    float r = 1.0f / s;
    #pragma unroll
    for (int u = 0; u < 8; ++u) row[lane + u * 32] = e[u] * r;
}

// ------- k_reps: masked copy of kx -------
__global__ void kreps_kernel(const float* __restrict__ kx,
                             const float* __restrict__ kmask,
                             float* __restrict__ out)
{
    long long idx = (long long)blockIdx.x * blockDim.x + threadIdx.x;
    const long long n = (long long)BB * MM * DIMD;
    if (idx >= n) return;
    int b = (int)(idx / (MM * DIMD));
    int m = (int)((idx / DIMD) % MM);
    out[idx] = (kmask[(size_t)b * MM + m] < 0.5f) ? 0.0f : kx[idx];
}

// ---------------- host side ----------------
static inline void launch_gemm(hipStream_t s,
    const float* A, const float* Bm, const float* bias, const float* Cres, float* C,
    int M, int N, int K, int lda, int ldb, int ldc,
    long long sAb, long long sAh, long long sBb, long long sBh,
    long long sCb, long long sCh, int Hdim, int batch,
    int transB, float alpha, int act)
{
    dim3 grid((N + BNT - 1) / BNT, (M + BMT - 1) / BMT, batch);
    gemm_wmma_bf16<<<grid, dim3(256), 0, s>>>(A, Bm, bias, Cres, C,
        M, N, K, lda, ldb, ldc, sAb, sAh, sBb, sBh, sCb, sCh,
        Hdim, transB, alpha, act);
}

extern "C" void kernel_launch(void* const* d_in, const int* in_sizes, int n_in,
                              void* d_out, int out_size, void* d_ws, size_t ws_size,
                              hipStream_t stream) {
    (void)in_sizes; (void)n_in; (void)out_size; (void)ws_size;
    const float* in_x    = (const float*)d_in[0];
    const float* in_kx   = (const float*)d_in[1];
    const float* in_rd   = (const float*)d_in[2];
    const float* in_clst = (const float*)d_in[3];
    const float* in_mask = (const float*)d_in[4];
    const float* in_kmsk = (const float*)d_in[5];
    const float* ln0_g   = (const float*)d_in[6];
    const float* ln0_b   = (const float*)d_in[7];
    const float* qkv_w   = (const float*)d_in[8];
    const float* out_w   = (const float*)d_in[9];
    const float* out_b   = (const float*)d_in[10];
    const float* ln1_g   = (const float*)d_in[11];
    const float* ln1_b   = (const float*)d_in[12];
    const float* ff_w1   = (const float*)d_in[13];
    const float* ff_b1   = (const float*)d_in[14];
    const float* ff_w2   = (const float*)d_in[15];
    const float* ff_b2   = (const float*)d_in[16];
    float* out = (float*)d_out;

    // workspace layout (floats)
    float* ws = (float*)d_ws;
    float* x_a   = ws;                       // 4,194,304
    float* x_b   = x_a + (size_t)BB*NN*DIMD; // 4,194,304
    float* kx_a  = x_b + (size_t)BB*NN*DIMD; // 262,144
    float* kx_b  = kx_a + (size_t)BB*MM*DIMD;
    float* c_a   = kx_b + (size_t)BB*MM*DIMD;   // 1,024
    float* c_b   = c_a + (size_t)BB*DIMD;
    float* qkv_x = c_b + (size_t)BB*DIMD;       // 12,582,912
    float* qkv_k = qkv_x + (size_t)BB*NN*THREE; // 786,432
    float* qkv_c = qkv_k + (size_t)BB*MM*THREE; // 3,072
    float* ao_x  = qkv_c + (size_t)BB*THREE;    // 4,194,304
    float* ao_k  = ao_x + (size_t)BB*NN*INNER;  // 262,144
    float* ao_c  = ao_k + (size_t)BB*MM*INNER;  // 1,024
    float* cdots = ao_c + (size_t)BB*INNER;     // 4,096
    float* ffh   = cdots + (size_t)BB*HEADS*MM; // 16,777,216
    float* S1    = ffh + (size_t)BB*NN*MLPD;    // 16,777,216

    // output layout
    float* out_kreps = out;                                    // DEPTH*B*M*DIM
    float* out_clst  = out + (size_t)DEPTH*BB*MM*DIMD;         // B*DIM
    float* out_amaps = out_clst + (size_t)BB*DIMD;             // DEPTH*B*H*M*N

    const float scale = 0.125f;   // 64^-0.5

    // init state from inputs (never mutate d_in)
    hipMemcpyAsync(x_a,  in_x,    (size_t)BB*NN*DIMD*sizeof(float), hipMemcpyDeviceToDevice, stream);
    hipMemcpyAsync(kx_a, in_kx,   (size_t)BB*MM*DIMD*sizeof(float), hipMemcpyDeviceToDevice, stream);
    hipMemcpyAsync(c_a,  in_clst, (size_t)BB*DIMD*sizeof(float),    hipMemcpyDeviceToDevice, stream);

    for (int l = 0; l < DEPTH; ++l) {
        const float* g0 = ln0_g + (size_t)l*DIMD;
        const float* b0 = ln0_b + (size_t)l*DIMD;
        const float* g1 = ln1_g + (size_t)l*DIMD;
        const float* b1 = ln1_b + (size_t)l*DIMD;
        const float* Wqkv = qkv_w + (size_t)l*DIMD*THREE;
        const float* Wout = out_w + (size_t)l*INNER*DIMD;
        const float* Bout = out_b + (size_t)l*DIMD;
        const float* W1 = ff_w1 + (size_t)l*DIMD*MLPD;
        const float* Bf1 = ff_b1 + (size_t)l*MLPD;
        const float* W2 = ff_w2 + (size_t)l*MLPD*DIMD;
        const float* Bf2 = ff_b2 + (size_t)l*DIMD;

        // ---- LN0 ----
        ln_kernel<<<dim3((BB*NN+7)/8), dim3(256), 0, stream>>>(x_a, g0, b0, x_b, BB*NN);
        ln_kernel<<<dim3((BB*MM+7)/8), dim3(256), 0, stream>>>(kx_a, g0, b0, kx_b, BB*MM);
        ln_kernel<<<dim3(1), dim3(256), 0, stream>>>(c_a, g0, b0, c_b, BB);

        // ---- QKV projections ----
        launch_gemm(stream, x_b, Wqkv, nullptr, nullptr, qkv_x,
                    BB*NN, THREE, DIMD, DIMD, THREE, THREE,
                    0,0,0,0,0,0, 1, 1, 0, 1.0f, 0);
        launch_gemm(stream, kx_b, Wqkv, nullptr, nullptr, qkv_k,
                    BB*MM, THREE, DIMD, DIMD, THREE, THREE,
                    0,0,0,0,0,0, 1, 1, 0, 1.0f, 0);
        launch_gemm(stream, c_b, Wqkv, nullptr, nullptr, qkv_c,
                    BB, THREE, DIMD, DIMD, THREE, THREE,
                    0,0,0,0,0,0, 1, 1, 0, 1.0f, 0);

        // ---- dots = t_q @ k_k^T * scale  -> S1 (B,H,N,M) ----
        launch_gemm(stream, qkv_x + 0, qkv_k + INNER, nullptr, nullptr, S1,
                    NN, MM, DHEAD, THREE, THREE, MM,
                    (long long)NN*THREE, DHEAD, (long long)MM*THREE, DHEAD,
                    (long long)HEADS*NN*MM, (long long)NN*MM,
                    HEADS, BB*HEADS, 1, scale, 0);

        // ---- softmax + *krd_t ; write amaps (transposed sharp) ----
        softmax_attn<<<dim3((BB*HEADS*NN)/8), dim3(256), 0, stream>>>(
            S1, in_rd, in_mask, in_kmsk,
            out_amaps + (size_t)l*BB*HEADS*MM*NN);

        // ---- att_out = attn @ k_v  (merged heads into ao_x) ----
        launch_gemm(stream, S1, qkv_k + 2*INNER, nullptr, nullptr, ao_x,
                    NN, DHEAD, MM, MM, THREE, INNER,
                    (long long)HEADS*NN*MM, (long long)NN*MM,
                    (long long)MM*THREE, DHEAD,
                    (long long)NN*INNER, DHEAD,
                    HEADS, BB*HEADS, 0, 1.0f, 0);

        // ---- k_dots = k_q @ t_k^T * scale -> S1 (B,H,M,N), reuse S1 ----
        launch_gemm(stream, qkv_k + 0, qkv_x + INNER, nullptr, nullptr, S1,
                    MM, NN, DHEAD, THREE, THREE, NN,
                    (long long)MM*THREE, DHEAD, (long long)NN*THREE, DHEAD,
                    (long long)HEADS*MM*NN, (long long)MM*NN,
                    HEADS, BB*HEADS, 1, scale, 0);

        // ---- k softmax * krd ----
        softmax_kattn<<<dim3(BB*HEADS*MM), dim3(256), 0, stream>>>(
            S1, in_rd, in_mask, in_kmsk);

        // ---- k_out = k_attn @ t_v -> ao_k ----
        launch_gemm(stream, S1, qkv_x + 2*INNER, nullptr, nullptr, ao_k,
                    MM, DHEAD, NN, NN, THREE, INNER,
                    (long long)HEADS*MM*NN, (long long)MM*NN,
                    (long long)NN*THREE, DHEAD,
                    (long long)MM*INNER, DHEAD,
                    HEADS, BB*HEADS, 0, 1.0f, 0);

        // ---- c_dots = c_q @ k_k^T * scale -> cdots (B,H,1,M) ----
        launch_gemm(stream, qkv_c + 0, qkv_k + INNER, nullptr, nullptr, cdots,
                    1, MM, DHEAD, THREE, THREE, MM,
                    (long long)THREE, DHEAD, (long long)MM*THREE, DHEAD,
                    (long long)HEADS*MM, MM,
                    HEADS, BB*HEADS, 1, scale, 0);
        softmax_c<<<dim3((BB*HEADS+7)/8), dim3(256), 0, stream>>>(cdots, in_mask, in_kmsk);
        // ---- c_out = c_attn @ k_v -> ao_c ----
        launch_gemm(stream, cdots, qkv_k + 2*INNER, nullptr, nullptr, ao_c,
                    1, DHEAD, MM, MM, THREE, INNER,
                    (long long)HEADS*MM, MM,
                    (long long)MM*THREE, DHEAD,
                    (long long)INNER, DHEAD,
                    HEADS, BB*HEADS, 0, 1.0f, 0);

        // ---- output projections + residual (residual = post-LN state) ----
        launch_gemm(stream, ao_x, Wout, Bout, x_b, x_a,
                    BB*NN, DIMD, INNER, INNER, DIMD, DIMD,
                    0,0,0,0,0,0, 1, 1, 0, 1.0f, 0);
        launch_gemm(stream, ao_k, Wout, Bout, kx_b, kx_a,
                    BB*MM, DIMD, INNER, INNER, DIMD, DIMD,
                    0,0,0,0,0,0, 1, 1, 0, 1.0f, 0);
        launch_gemm(stream, ao_c, Wout, Bout, c_b, c_a,
                    BB, DIMD, INNER, INNER, DIMD, DIMD,
                    0,0,0,0,0,0, 1, 1, 0, 1.0f, 0);

        // ---- FF blocks: t = gelu(LN(t)@W1+b1)@W2+b2 + t ----
        ln_kernel<<<dim3((BB*NN+7)/8), dim3(256), 0, stream>>>(x_a, g1, b1, x_b, BB*NN);
        launch_gemm(stream, x_b, W1, Bf1, nullptr, ffh,
                    BB*NN, MLPD, DIMD, DIMD, MLPD, MLPD,
                    0,0,0,0,0,0, 1, 1, 0, 1.0f, 1);
        launch_gemm(stream, ffh, W2, Bf2, x_a, x_a,
                    BB*NN, DIMD, MLPD, MLPD, DIMD, DIMD,
                    0,0,0,0,0,0, 1, 1, 0, 1.0f, 0);

        ln_kernel<<<dim3((BB*MM+7)/8), dim3(256), 0, stream>>>(kx_a, g1, b1, kx_b, BB*MM);
        launch_gemm(stream, kx_b, W1, Bf1, nullptr, ffh,
                    BB*MM, MLPD, DIMD, DIMD, MLPD, MLPD,
                    0,0,0,0,0,0, 1, 1, 0, 1.0f, 1);
        launch_gemm(stream, ffh, W2, Bf2, kx_a, kx_a,
                    BB*MM, DIMD, MLPD, MLPD, DIMD, DIMD,
                    0,0,0,0,0,0, 1, 1, 0, 1.0f, 0);

        ln_kernel<<<dim3(1), dim3(256), 0, stream>>>(c_a, g1, b1, c_b, BB);
        launch_gemm(stream, c_b, W1, Bf1, nullptr, ffh,
                    BB, MLPD, DIMD, DIMD, MLPD, MLPD,
                    0,0,0,0,0,0, 1, 1, 0, 1.0f, 1);
        launch_gemm(stream, ffh, W2, Bf2, c_a, c_a,
                    BB, DIMD, MLPD, MLPD, DIMD, DIMD,
                    0,0,0,0,0,0, 1, 1, 0, 1.0f, 0);

        // ---- k_reps[l] = where(kmask<0.5, 0, kx) ----
        {
            long long n = (long long)BB*MM*DIMD;
            kreps_kernel<<<dim3((unsigned)((n + 255) / 256)), dim3(256), 0, stream>>>(
                kx_a, in_kmsk, out_kreps + (size_t)l*BB*MM*DIMD);
        }
    }

    // final clst output
    hipMemcpyAsync(out_clst, c_a, (size_t)BB*DIMD*sizeof(float),
                   hipMemcpyDeviceToDevice, stream);
}
